// MoELayer_37976100831417
// MI455X (gfx1250) — matrix operand order repaired and checked by reading, not compiled, not run on hardware
//
#include <hip/hip_runtime.h>
#include <cstdint>

// ---------------- types ----------------
typedef __bf16 bf16_t;
typedef __attribute__((ext_vector_type(16))) __bf16 v16bf;
typedef __attribute__((ext_vector_type(8)))  __bf16 v8bf;
typedef __attribute__((ext_vector_type(8)))  float  v8f;

// ---------------- problem constants ----------------
constexpr int N = 16384, D = 1024, E = 8, H = 4096;

// LDS layout for the fused expert kernel (dynamic shared memory)
constexpr int XG_LD = D + 8;                                     // bf16 elems, row stride (bank-skew)
constexpr int H_LD  = H + 8;
constexpr size_t XG_BYTES = (size_t)16 * XG_LD * sizeof(bf16_t); // 33,024 B
constexpr size_t HB_BYTES = (size_t)16 * H_LD  * sizeof(bf16_t); // 131,328 B
constexpr size_t SMEM_MOE = XG_BYTES + HB_BYTES + 64 + 64;       // 164,480 B (< 320KB WGP LDS)

// ---------------- fragment helpers (CDNA5 WMMA 16x16x32 bf16) ----------------
// A 16x32 (MxK), 16-bit: lane<16 -> row=lane, elems 0..15 = K(kb+0..15)
//                        lane>=16 -> row=lane-16, elems 0..15 = K(kb+16..31).
__device__ __forceinline__ v16bf load_a_frag(const bf16_t* row, int kbase, int kh) {
  v8bf lo = *(const v8bf*)(row + kbase + kh * 16);
  v8bf hi = *(const v8bf*)(row + kbase + kh * 16 + 8);
  v16bf r;
#pragma unroll
  for (int i = 0; i < 8; ++i) { r[i] = lo[i]; r[i + 8] = hi[i]; }
  return r;
}
// B 32x16 (KxN), 16-bit: lane holds col = lane&15, K = (lane>>4)*16 + i (i=0..15)
// With K contiguous in memory (transposed weights) this is one 32B read per lane.
__device__ __forceinline__ v16bf load_b_frag(const bf16_t* p) {
  v8bf lo = *(const v8bf*)p;
  v8bf hi = *(const v8bf*)(p + 8);
  v16bf r;
#pragma unroll
  for (int i = 0; i < 8; ++i) { r[i] = lo[i]; r[i + 8] = hi[i]; }
  return r;
}

// ---------------- kernel 1: init counters ----------------
__global__ void init_kernel(int* counts, int* cursors, float* sumg) {
  int i = threadIdx.x;
  if (i < E) { counts[i] = 0; cursors[i] = 0; }
  if (i == 0) sumg[0] = 0.f;
}

// ---------------- kernel 2/3: cast fp32 -> bf16 with transpose (per expert) ----
// src: E matrices of R x C (row-major) -> dst: E matrices of C x R (row-major)
__global__ void transpose_cast_kernel(const float* __restrict__ src,
                                      bf16_t* __restrict__ dst, int R, int C) {
  __shared__ float tile[32][33];
  const size_t mat = (size_t)R * C;
  src += (size_t)blockIdx.z * mat;
  dst += (size_t)blockIdx.z * mat;
  int c0 = blockIdx.x * 32, r0 = blockIdx.y * 32;
#pragma unroll
  for (int i = threadIdx.y; i < 32; i += 8)
    tile[i][threadIdx.x] = src[(size_t)(r0 + i) * C + c0 + threadIdx.x];
  __syncthreads();
#pragma unroll
  for (int i = threadIdx.y; i < 32; i += 8)
    dst[(size_t)(c0 + i) * R + r0 + threadIdx.x] = (bf16_t)tile[threadIdx.x][i];
}

// ---------------- kernel 4: cast x -> bf16 ----------------
__global__ void cast_kernel(const float* __restrict__ src, bf16_t* __restrict__ dst, int n) {
  int i = blockIdx.x * blockDim.x + threadIdx.x;
  if (i < n) dst[i] = (bf16_t)src[i];
}

// ---------------- kernel 5: gate (softmax + top2 + renorm + counts) ----------
__global__ void gate_kernel(const float* __restrict__ x, const float* __restrict__ Wg,
                            const float* __restrict__ bg, int* __restrict__ counts,
                            int* __restrict__ toke, float* __restrict__ tokg,
                            float* __restrict__ sumg) {
  int n = blockIdx.x * 256 + threadIdx.x;
  float acc[E];
#pragma unroll
  for (int e = 0; e < E; ++e) acc[e] = bg[e];
  const float4* x4 = (const float4*)(x + (size_t)n * D);
  for (int d4 = 0; d4 < D / 4; ++d4) {
    float4 xv = x4[d4];
    const float* w = Wg + (size_t)d4 * 4 * E;
#pragma unroll
    for (int e = 0; e < E; ++e)
      acc[e] += xv.x * w[e] + xv.y * w[e + E] + xv.z * w[e + 2 * E] + xv.w * w[e + 3 * E];
  }
  float mx = acc[0];
#pragma unroll
  for (int e = 1; e < E; ++e) mx = fmaxf(mx, acc[e]);
  float p[E], s = 0.f;
#pragma unroll
  for (int e = 0; e < E; ++e) { p[e] = expf(acc[e] - mx); s += p[e]; }
  int i0 = 0, i1 = -1;
  float b0 = p[0], b1v = -1.f;
#pragma unroll
  for (int e = 1; e < E; ++e) {
    if (p[e] > b0) { b1v = b0; i1 = i0; b0 = p[e]; i0 = e; }
    else if (p[e] > b1v) { b1v = p[e]; i1 = e; }
  }
  float g0 = b0 / s, g1 = b1v / s;      // top-2 softmax probs
  float gs = g0 + g1;
  g0 /= gs; g1 /= gs;                    // renormalize
  toke[n * 2] = i0; toke[n * 2 + 1] = i1;
  tokg[n * 2] = g0; tokg[n * 2 + 1] = g1;
  atomicAdd(&counts[i0], 1);
  atomicAdd(&counts[i1], 1);
  // block-reduce mean(gates[:,0]) contribution
  __shared__ float sred[256];
  sred[threadIdx.x] = g0;
  __syncthreads();
  for (int st = 128; st > 0; st >>= 1) {
    if (threadIdx.x < st) sred[threadIdx.x] += sred[threadIdx.x + st];
    __syncthreads();
  }
  if (threadIdx.x == 0) atomicAdd(sumg, sred[0]);
}

// ---------------- kernel 6: exclusive prefix over E=8 counts ----------------
__global__ void prefix_kernel(const int* __restrict__ counts, int* __restrict__ offsets) {
  if (threadIdx.x == 0) {
    int a = 0;
#pragma unroll
    for (int e = 0; e < E; ++e) { offsets[e] = a; a += counts[e]; }
    offsets[E] = a;
  }
}

// ---------------- kernel 7: scatter tokens into per-expert lists -------------
__global__ void scatter_kernel(const int* __restrict__ toke, const float* __restrict__ tokg,
                               const int* __restrict__ offsets, int* __restrict__ cursors,
                               int* __restrict__ listt, float* __restrict__ listg,
                               int* __restrict__ slots) {
  int n = blockIdx.x * 256 + threadIdx.x;
  if (n >= N) return;
#pragma unroll
  for (int k = 0; k < 2; ++k) {
    int e = toke[n * 2 + k];
    int pos = atomicAdd(&cursors[e], 1);
    int sl = offsets[e] + pos;
    listt[sl] = n;
    listg[sl] = tokg[n * 2 + k];
    slots[n * 2 + k] = sl;
  }
}

// ---------------- kernel 8: fused 2-layer expert MLP (WMMA bf16) -------------
// block = 256 threads (8 waves). One block = 16 gathered tokens x 1 expert.
// GEMM1: (16xD)x(DxH) +b1, ReLU  -> h in LDS (bf16)
// GEMM2: (16xH)x(HxD) +b2, *gate -> ybuf[slot]
__global__ void __launch_bounds__(256)
moe_expert_kernel(const bf16_t* __restrict__ xb, const bf16_t* __restrict__ W1T,
                  const bf16_t* __restrict__ W2T, const float* __restrict__ b1,
                  const float* __restrict__ b2, const int* __restrict__ counts,
                  const int* __restrict__ offsets, const int* __restrict__ listt,
                  const float* __restrict__ listg, float* __restrict__ ybuf) {
  const int e = blockIdx.x >> 10;       // E * (N/16) blocks
  const int t = blockIdx.x & 1023;
  const int cnt = counts[e];
  if (t * 16 >= cnt) return;            // routed: skip empty tiles
  const int off_e = offsets[e];

  extern __shared__ char smem[];
  bf16_t* xg   = (bf16_t*)smem;                         // [16][XG_LD]
  bf16_t* hbuf = (bf16_t*)(smem + XG_BYTES);            // [16][H_LD]
  int*    s_tok  = (int*)(smem + XG_BYTES + HB_BYTES);
  float*  s_gate = (float*)(smem + XG_BYTES + HB_BYTES + 64);

  const int lane = threadIdx.x & 31;
  const int wave = threadIdx.x >> 5;
  const int kh   = lane >> 4;           // half-select (A K-half / B K-half / C row-half)
  const int colh = lane & 15;           // A row / B col / C col

  if (threadIdx.x < 16) {
    int gpos = t * 16 + threadIdx.x;
    if (gpos < cnt) { s_tok[threadIdx.x] = listt[off_e + gpos]; s_gate[threadIdx.x] = listg[off_e + gpos]; }
    else            { s_tok[threadIdx.x] = listt[off_e];        s_gate[threadIdx.x] = 0.f; }
  }
  __syncthreads();

  // gather 16 token rows of x (bf16) into LDS, 16B chunks
  for (int c = threadIdx.x; c < 16 * (D / 8); c += 256) {
    int row = c / (D / 8), o = (c % (D / 8)) * 8;
    *(uint4*)(xg + row * XG_LD + o) = *(const uint4*)(xb + (size_t)s_tok[row] * D + o);
  }
  __syncthreads();

  // ---- GEMM1: h = relu(xg @ W1[e] + b1[e]) ----
  const bf16_t* xrow = xg + (size_t)colh * XG_LD;
  const bf16_t* w1e  = W1T + (size_t)e * D * H;   // [H][D]
  const float*  b1e  = b1 + (size_t)e * H;
#pragma unroll 1
  for (int p = 0; p < 32; p += 2) {               // 32 N-tiles per wave, paired for ILP
    const int n0 = (wave * 32 + p) * 16, n1 = n0 + 16;
    const bf16_t* bp0 = w1e + (size_t)(n0 + colh) * D + kh * 16;
    const bf16_t* bp1 = w1e + (size_t)(n1 + colh) * D + kh * 16;
    v8f acc0 = {}, acc1 = {};
#pragma unroll 4
    for (int kt = 0; kt < D / 32; ++kt) {
      v16bf a  = load_a_frag(xrow, kt * 32, kh);
      v16bf f0 = load_b_frag(bp0 + kt * 32);
      v16bf f1 = load_b_frag(bp1 + kt * 32);
      acc0 = __builtin_amdgcn_wmma_f32_16x16x32_bf16(false, a, false, f0, (short)0, acc0, false, false);
      acc1 = __builtin_amdgcn_wmma_f32_16x16x32_bf16(false, a, false, f1, (short)0, acc1, false, false);
    }
    const float bias0 = b1e[n0 + colh], bias1 = b1e[n1 + colh];
#pragma unroll
    for (int j = 0; j < 8; ++j) {
      int r = j + 8 * kh;
      hbuf[r * H_LD + n0 + colh] = (bf16_t)fmaxf(acc0[j] + bias0, 0.f);
      hbuf[r * H_LD + n1 + colh] = (bf16_t)fmaxf(acc1[j] + bias1, 0.f);
    }
  }
  __syncthreads();

  // ---- GEMM2: y = gate * (h @ W2[e] + b2[e]) -> ybuf[slot] ----
  const bf16_t* hrow = hbuf + (size_t)colh * H_LD;
  const bf16_t* w2e  = W2T + (size_t)e * H * D;   // [D][H]
  const float*  b2e  = b2 + (size_t)e * D;
#pragma unroll 1
  for (int p = 0; p < 8; p += 2) {                // 8 N-tiles per wave, paired
    const int d0 = (wave * 8 + p) * 16, d1 = d0 + 16;
    const bf16_t* bp0 = w2e + (size_t)(d0 + colh) * H + kh * 16;
    const bf16_t* bp1 = w2e + (size_t)(d1 + colh) * H + kh * 16;
    v8f acc0 = {}, acc1 = {};
#pragma unroll 4
    for (int kt = 0; kt < H / 32; ++kt) {
      v16bf a  = load_a_frag(hrow, kt * 32, kh);
      v16bf f0 = load_b_frag(bp0 + kt * 32);
      v16bf f1 = load_b_frag(bp1 + kt * 32);
      acc0 = __builtin_amdgcn_wmma_f32_16x16x32_bf16(false, a, false, f0, (short)0, acc0, false, false);
      acc1 = __builtin_amdgcn_wmma_f32_16x16x32_bf16(false, a, false, f1, (short)0, acc1, false, false);
    }
    const float bv0 = b2e[d0 + colh], bv1 = b2e[d1 + colh];
#pragma unroll
    for (int j = 0; j < 8; ++j) {
      int r = j + 8 * kh;
      int gpos = t * 16 + r;
      if (gpos < cnt) {                          // skip pad rows -> deterministic, no atomics
        size_t srow = (size_t)(off_e + gpos) * D;
        float g = s_gate[r];
        ybuf[srow + d0 + colh] = g * (acc0[j] + bv0);
        ybuf[srow + d1 + colh] = g * (acc1[j] + bv1);
      }
    }
  }
}

// ---------------- kernel 9: combine the two routed contributions -------------
__global__ void combine_kernel(const float* __restrict__ ybuf, const int* __restrict__ slots,
                               float* __restrict__ out) {
  int idx = blockIdx.x * 256 + threadIdx.x;
  if (idx >= N * D) return;
  int n = idx >> 10, d = idx & (D - 1);
  out[idx] = ybuf[(size_t)slots[n * 2] * D + d] + ybuf[(size_t)slots[n * 2 + 1] * D + d];
}

// ---------------- kernel 10: aux loss ----------------
__global__ void aux_kernel(const int* __restrict__ counts, const float* __restrict__ sumg,
                           float* __restrict__ out_aux) {
  int act = 0;
#pragma unroll
  for (int e = 0; e < E; ++e) act += (counts[e] > 0) ? 1 : 0;
  float m = sumg[0] / (float)N;
  out_aux[0] = (float)act * m * m;
}

// ---------------- host launch ----------------
extern "C" void kernel_launch(void* const* d_in, const int* in_sizes, int n_in,
                              void* d_out, int out_size, void* d_ws, size_t ws_size,
                              hipStream_t stream) {
  (void)in_sizes; (void)n_in; (void)out_size; (void)ws_size;
  const float* x  = (const float*)d_in[0];
  const float* Wg = (const float*)d_in[1];
  const float* bg = (const float*)d_in[2];
  const float* W1 = (const float*)d_in[3];
  const float* b1 = (const float*)d_in[4];
  const float* W2 = (const float*)d_in[5];
  const float* b2 = (const float*)d_in[6];
  float* out = (float*)d_out;

  // workspace layout (256B aligned slices)
  char* ws = (char*)d_ws;
  size_t off = 0;
  auto take = [&](size_t bytes) -> void* {
    void* p = ws + off;
    off = (off + bytes + 255) & ~(size_t)255;
    return p;
  };
  int*    w_counts  = (int*)   take(E * sizeof(int));
  int*    w_cursors = (int*)   take(E * sizeof(int));
  int*    w_offsets = (int*)   take((E + 1) * sizeof(int));
  float*  w_sumg    = (float*) take(sizeof(float));
  int*    w_toke    = (int*)   take((size_t)2 * N * sizeof(int));
  float*  w_tokg    = (float*) take((size_t)2 * N * sizeof(float));
  int*    w_slots   = (int*)   take((size_t)2 * N * sizeof(int));
  int*    w_listt   = (int*)   take((size_t)2 * N * sizeof(int));
  float*  w_listg   = (float*) take((size_t)2 * N * sizeof(float));
  bf16_t* w_xb      = (bf16_t*)take((size_t)N * D * sizeof(bf16_t));
  bf16_t* w_w1t     = (bf16_t*)take((size_t)E * D * H * sizeof(bf16_t));
  bf16_t* w_w2t     = (bf16_t*)take((size_t)E * D * H * sizeof(bf16_t));
  float*  w_ybuf    = (float*) take((size_t)2 * N * D * sizeof(float));

  init_kernel<<<1, 256, 0, stream>>>(w_counts, w_cursors, w_sumg);
  // W1 (E,D,H) -> W1T (E,H,D)
  transpose_cast_kernel<<<dim3(H / 32, D / 32, E), dim3(32, 8), 0, stream>>>(W1, w_w1t, D, H);
  // W2 (E,H,D) -> W2T (E,D,H)
  transpose_cast_kernel<<<dim3(D / 32, H / 32, E), dim3(32, 8), 0, stream>>>(W2, w_w2t, H, D);
  cast_kernel<<<(N * D + 255) / 256, 256, 0, stream>>>(x, w_xb, N * D);
  gate_kernel<<<N / 256, 256, 0, stream>>>(x, Wg, bg, w_counts, w_toke, w_tokg, w_sumg);
  prefix_kernel<<<1, 32, 0, stream>>>(w_counts, w_offsets);
  scatter_kernel<<<N / 256, 256, 0, stream>>>(w_toke, w_tokg, w_offsets, w_cursors,
                                              w_listt, w_listg, w_slots);
  moe_expert_kernel<<<E * (N / 16), 256, SMEM_MOE, stream>>>(
      w_xb, w_w1t, w_w2t, b1, b2, w_counts, w_offsets, w_listt, w_listg, w_ybuf);
  combine_kernel<<<(N * D) / 256, 256, 0, stream>>>(w_ybuf, w_slots, out);
  aux_kernel<<<1, 1, 0, stream>>>(w_counts, w_sumg, out + (size_t)N * D);
}
